// LSTM_28827820490953
// MI455X (gfx1250) — matrix operand order repaired and checked by reading, not compiled, not run on hardware
//
#include <hip/hip_runtime.h>
#include <hip/hip_bf16.h>

typedef __attribute__((ext_vector_type(16))) __bf16 v16bf;
typedef __attribute__((ext_vector_type(8)))  float  v8f;

#define T_STEPS 256
#define NB      256          // batch
#define DH      512          // D == H == 512
#define G4      2048         // 4*H
#define MAT_ELEMS (DH * G4)  // 1,048,576 elements per weight matrix

// ---------------------------------------------------------------------------
// Weight repack: fp32 row-major [K=512][4H=2048]  ->  bf16 WMMA B-fragments.
// Fragment (nt, kt) covers cols nt*16..+15, K kt*32..+31 (512 bf16 = 1 KB).
// Within a fragment, lane l's 16 values are contiguous (one 32B load):
//   element j -> K = kt*32 + (j>=8 ? 16 : 0) + (l>=16 ? 8 : 0) + (j&7), col = nt*16 + (l&15)
// (16-bit B layout: lanes 0-15 hold K {0..7,16..23}, lanes 16-31 hold K {8..15,24..31},
//  matching the A-fragment K order.)
// Matrices packed in order: Wi[0], Wh[0], Wi[1], Wh[1].
// ---------------------------------------------------------------------------
__global__ __launch_bounds__(256) void pack_w(const float* __restrict__ Wi,
                                              const float* __restrict__ Wh,
                                              __bf16* __restrict__ packed) {
  unsigned gid  = blockIdx.x * 256u + threadIdx.x;   // 4 * MAT_ELEMS total
  unsigned mm   = gid >> 20;                         // matrix 0..3
  unsigned e    = gid & (MAT_ELEMS - 1u);
  unsigned frag = e >> 9;                            // nt*16 + kt
  unsigned win  = e & 511u;
  unsigned l    = win >> 4, j = win & 15u;
  unsigned nt   = frag >> 4, kt = frag & 15u;
  unsigned K    = kt * 32u + ((j & 8u) ? 16u : 0u) + ((l & 16u) ? 8u : 0u) + (j & 7u);
  unsigned col  = nt * 16u + (l & 15u);
  const float* src = ((mm & 1u) ? Wh : Wi) + (mm >> 1) * (size_t)MAT_ELEMS;
  packed[gid] = (__bf16)src[(size_t)K * G4 + col];
}

// ---------------------------------------------------------------------------
// Copy start_c / start_h into ping-pong buffer slot 0 of each layer.
// buffers laid out as [layer][pingpong][NB*DH]
// ---------------------------------------------------------------------------
__global__ __launch_bounds__(256) void init_state(const float* __restrict__ start_c,
                                                  const float* __restrict__ start_h,
                                                  float* __restrict__ cbuf,
                                                  float* __restrict__ hbuf) {
  unsigned e = blockIdx.x * 256u + threadIdx.x;      // < 2*NB*DH = 262144
  unsigned l   = e >> 17;                            // NB*DH = 131072 = 1<<17
  unsigned off = e & 131071u;
  cbuf[(l * 2u) * 131072u + off] = start_c[e];
  hbuf[(l * 2u) * 131072u + off] = start_h[e];
}

// ---------------------------------------------------------------------------
// Build a 16x32 bf16 A-fragment from a row-major fp32 activation row.
// 16-bit A layout (CDNA5 ISA): lane<16 holds K {k0..k0+7, k0+16..k0+23},
// lane>=16 holds K {k0+8..k0+15, k0+24..k0+31}; row = m0 + (lane&15).
// `scale` (1.0 or 0.0) implements the episode-break carry mask for h.
// ---------------------------------------------------------------------------
__device__ __forceinline__ v16bf load_a_frag(const float* __restrict__ row,
                                             int k0, int lane, float scale) {
  const int hi = (lane & 16) ? 8 : 0;
  const float4 f0 = *(const float4*)(row + k0 + hi);
  const float4 f1 = *(const float4*)(row + k0 + hi + 4);
  const float4 f2 = *(const float4*)(row + k0 + 16 + hi);
  const float4 f3 = *(const float4*)(row + k0 + 16 + hi + 4);
  v16bf a;
  a[0]  = (__bf16)(f0.x * scale); a[1]  = (__bf16)(f0.y * scale);
  a[2]  = (__bf16)(f0.z * scale); a[3]  = (__bf16)(f0.w * scale);
  a[4]  = (__bf16)(f1.x * scale); a[5]  = (__bf16)(f1.y * scale);
  a[6]  = (__bf16)(f1.z * scale); a[7]  = (__bf16)(f1.w * scale);
  a[8]  = (__bf16)(f2.x * scale); a[9]  = (__bf16)(f2.y * scale);
  a[10] = (__bf16)(f2.z * scale); a[11] = (__bf16)(f2.w * scale);
  a[12] = (__bf16)(f3.x * scale); a[13] = (__bf16)(f3.y * scale);
  a[14] = (__bf16)(f3.z * scale); a[15] = (__bf16)(f3.w * scale);
  return a;
}

__device__ __forceinline__ float tanh_fast(float x) {
  float e = __expf(2.f * x);
  return 1.f - 2.f / (e + 1.f);
}

// ---------------------------------------------------------------------------
// One LSTM cell step for one layer:
//   z = x @ WiP + h_prev(masked) @ WhP + bias ;  gates ;  write c,h,out
// Grid: (8 unit-groups of 64, 4 row-groups of 64), 256 threads = 8 waves.
// Wave w: row-strip rs=w&3 (16 rows), column-half ch=w>>2 (8 of 16 col-tiles).
// Column tiles ct=0..15 map to (gate = ct>>2, unit-subtile = ct&3) so the
// block owns all four gates for its 64 units; z staged in LDS for gating.
// ---------------------------------------------------------------------------
__global__ __launch_bounds__(256) void lstm_step(
    const float* __restrict__ Ax,          // [NB][DH] x input (fp32 row-major)
    const __bf16* __restrict__ BxP,        // packed Wi for this layer
    const float* __restrict__ Ah,          // [NB][DH] h_prev
    const __bf16* __restrict__ BhP,        // packed Wh for this layer
    const float* __restrict__ bias,        // [G4]
    const unsigned char* __restrict__ brk, // [NB] episode-break flags
    const float* __restrict__ c_prev,      // [NB][DH]
    float* __restrict__ c_out,
    float* __restrict__ h_out,
    float* __restrict__ outp)              // out slice, row stride 1024
{
  __shared__ float zsh[64 * 256];          // 64 rows x 256 gate-cols, 64 KB

  const int tid  = threadIdx.x;
  const int lane = tid & 31;
  const int w    = tid >> 5;
  const int rs   = w & 3;
  const int ch   = w >> 2;
  const int u0   = blockIdx.x * 64;        // unit base
  const int m0g  = blockIdx.y * 64;        // row base
  const int m0   = m0g + rs * 16;
  const int row_a = m0 + (lane & 15);

  v8f acc[8] = {};

  int ntg[8];                              // global 16-col tile index per tile
#pragma unroll
  for (int tt = 0; tt < 8; ++tt) {
    const int ct   = ch * 8 + tt;
    const int gate = ct >> 2, usub = ct & 3;
    ntg[tt] = gate * 32 + (u0 >> 4) + usub;
  }

  const float hscale = brk[row_a] ? 0.f : 1.f;

  // phase 0: x @ Wi (unmasked) ; phase 1: h_prev @ Wh (break-masked)
#pragma unroll 1
  for (int phase = 0; phase < 2; ++phase) {
    const float*  Arow = (phase ? Ah : Ax) + (size_t)row_a * DH;
    const __bf16* Bp   = phase ? BhP : BxP;
    const float   sc   = phase ? hscale : 1.0f;
#pragma unroll 1
    for (int kt = 0; kt < 16; ++kt) {      // K = 512 in chunks of 32
      v16bf a = load_a_frag(Arow, kt * 32, lane, sc);
#pragma unroll
      for (int tt = 0; tt < 8; ++tt) {
        v16bf bfr = *(const v16bf*)(Bp + ((size_t)(ntg[tt] * 16 + kt) * 512)
                                        + lane * 16);
        acc[tt] = __builtin_amdgcn_wmma_f32_16x16x32_bf16(
            false, a, false, bfr, (short)0, acc[tt], false, false);
      }
    }
  }

  // scatter accumulators to LDS: D element r -> (M = r + (lane>=16?8:0), N = lane&15)
  const int hi8 = (lane & 16) ? 8 : 0;
  const int nl  = lane & 15;
#pragma unroll
  for (int tt = 0; tt < 8; ++tt) {
    const int ct   = ch * 8 + tt;
    const int gate = ct >> 2, usub = ct & 3;
    const int c_lcl = gate * 64 + usub * 16 + nl;
#pragma unroll
    for (int r = 0; r < 8; ++r)
      zsh[(rs * 16 + hi8 + r) * 256 + c_lcl] = acc[tt][r];
  }
  __syncthreads();

  // elementwise gates: 64 rows x 64 units = 4096 elems, 16 per thread
#pragma unroll 1
  for (int i = 0; i < 16; ++i) {
    const int idx  = i * 256 + tid;
    const int u    = idx & 63;
    const int mloc = idx >> 6;
    const int rg   = m0g + mloc;
    const int ug   = u0 + u;
    const float gi = zsh[mloc * 256 +   0 + u] + bias[ug];
    const float gf = zsh[mloc * 256 +  64 + u] + bias[512 + ug];
    const float gg = zsh[mloc * 256 + 128 + u] + bias[1024 + ug];
    const float go = zsh[mloc * 256 + 192 + u] + bias[1536 + ug];
    const float mc = brk[rg] ? 0.f : c_prev[(size_t)rg * DH + ug];
    const float si = 1.f / (1.f + __expf(-gi));
    const float sf = 1.f / (1.f + __expf(-gf));
    const float so = 1.f / (1.f + __expf(-go));
    const float nc = sf * mc + si * tanh_fast(gg);
    const float nh = so * tanh_fast(nc);
    c_out[(size_t)rg * DH + ug] = nc;
    h_out[(size_t)rg * DH + ug] = nh;
    outp[(size_t)rg * 1024 + ug] = nh;
  }
}

// ---------------------------------------------------------------------------
// Host side: pack weights once, init state, then 2 kernels per timestep
// (layer-1 consumes layer-0's freshly written h via stream ordering).
// Workspace: [0,8MB) packed bf16 weights; [8,10MB) h ping-pong; [10,12MB) c.
// ---------------------------------------------------------------------------
extern "C" void kernel_launch(void* const* d_in, const int* in_sizes, int n_in,
                              void* d_out, int out_size, void* d_ws, size_t ws_size,
                              hipStream_t stream) {
  const float* seq_x   = (const float*)d_in[0];
  const float* start_c = (const float*)d_in[1];
  const float* start_h = (const float*)d_in[2];
  const float* Wi      = (const float*)d_in[3];
  const float* Wh      = (const float*)d_in[4];
  const float* b       = (const float*)d_in[5];
  const unsigned char* brks = (const unsigned char*)d_in[6]; // bool[T,N,1]
  float* out = (float*)d_out;

  char* ws = (char*)d_ws;
  __bf16* packed = (__bf16*)ws;                               // 8 MB
  float*  hbuf   = (float*)(ws + (size_t)8  * 1024 * 1024);   // [2][2][131072]
  float*  cbuf   = (float*)(ws + (size_t)10 * 1024 * 1024);   // [2][2][131072]

  pack_w<<<(4 * MAT_ELEMS) / 256, 256, 0, stream>>>(Wi, Wh, packed);
  init_state<<<(2 * NB * DH) / 256, 256, 0, stream>>>(start_c, start_h, cbuf, hbuf);

  const dim3 grid(8, 4);
  for (int t = 0; t < T_STEPS; ++t) {
    const int src = t & 1, dst = src ^ 1;
    const unsigned char* brk_t = brks + (size_t)t * NB;
    float* out_t = out + (size_t)t * NB * 1024;

    // layer 0: x = seq_x[t]
    lstm_step<<<grid, 256, 0, stream>>>(
        seq_x + (size_t)t * NB * DH,
        packed + 0 * (size_t)MAT_ELEMS,                 // Wi[0]
        hbuf + (size_t)(0 * 2 + src) * NB * DH,
        packed + 1 * (size_t)MAT_ELEMS,                 // Wh[0]
        b + 0 * G4, brk_t,
        cbuf + (size_t)(0 * 2 + src) * NB * DH,
        cbuf + (size_t)(0 * 2 + dst) * NB * DH,
        hbuf + (size_t)(0 * 2 + dst) * NB * DH,
        out_t + 0 * DH);

    // layer 1: x = h0 (just written this step)
    lstm_step<<<grid, 256, 0, stream>>>(
        hbuf + (size_t)(0 * 2 + dst) * NB * DH,
        packed + 2 * (size_t)MAT_ELEMS,                 // Wi[1]
        hbuf + (size_t)(1 * 2 + src) * NB * DH,
        packed + 3 * (size_t)MAT_ELEMS,                 // Wh[1]
        b + 1 * G4, brk_t,
        cbuf + (size_t)(1 * 2 + src) * NB * DH,
        cbuf + (size_t)(1 * 2 + dst) * NB * DH,
        hbuf + (size_t)(1 * 2 + dst) * NB * DH,
        out_t + 1 * DH);
  }
}
